// HIVEMixer_75453985456635
// MI455X (gfx1250) — matrix operand order repaired and checked by reading, not compiled, not run on hardware
//
#include <hip/hip_runtime.h>
#include <hip/hip_bf16.h>
#include <math.h>

typedef __attribute__((ext_vector_type(16))) _Float16 v16h;
typedef __attribute__((ext_vector_type(8)))  float    v8f;
typedef __attribute__((ext_vector_type(4)))  unsigned int u32x4;
typedef __attribute__((ext_vector_type(8)))  int      i32x8;
typedef __attribute__((ext_vector_type(4)))  int      i32x4;

// Problem constants
#define NB   128      // BSZ*SL
#define NS   32       // SS
#define NAg  16
#define NEd  16
#define DD   64
#define SDm  128
#define HH   64
#define FFd  2048
#define NLr  4

// --- TDM availability / arity detection -------------------------------------
#if defined(__has_builtin)
#if __has_builtin(__builtin_amdgcn_tensor_load_to_lds)
#if __has_include(<hip/amd_detail/amd_gfx1250_TDM.h>)
#define TDM_MODE 6          // amdgpu-toolchain clang-23: 6-arg builtin
#else
#define TDM_MODE 5          // ROCm 7.2 clang-22: 5-arg builtin
#endif
#endif
#endif
#ifndef TDM_MODE
#define TDM_MODE 0
#endif

__device__ __forceinline__ v8f wmma_f16(v16h a, v16h b, v8f c) {
  return __builtin_amdgcn_wmma_f32_16x16x32_f16(false, a, false, b, (short)0, c,
                                                false, false);
}

// A fragment (16xK tile, f16 in LDS, row-major leading dim ld), K-offset k0.
// A layout (matches B): lanes 0-15 hold row M=lane, K=0..15;
// lanes 16-31 hold row M=lane-16, K=16..31.
__device__ __forceinline__ v16h frag_a_lds(const _Float16* A, int ld, int k0, int lane) {
  const int row = lane & 15;
  const int hb  = (lane >> 4) << 4;
  v16h a;
#pragma unroll
  for (int e = 0; e < 16; ++e) {
    int kk = e + hb;
    a[e] = A[row * ld + k0 + kk];
  }
  return a;
}

// B fragment (32x16 tile) from an f16 matrix [K x N] row-major.
// Lanes 0-15: col N=lane, K=e; lanes 16-31: K=e+16.
__device__ __forceinline__ v16h frag_b_h(const _Float16* B, int ldb, int k0, int n0, int lane) {
  const int col = lane & 15;
  const int hb  = (lane >> 4) << 4;
  v16h b;
#pragma unroll
  for (int e = 0; e < 16; ++e)
    b[e] = B[(k0 + e + hb) * ldb + n0 + col];
  return b;
}

// Generic LDS pointer -> 32-bit LDS byte address (LDS aperture keeps the
// offset in the low 32 bits of the generic address).
__device__ __forceinline__ unsigned lds_addr32(const void* p) {
  return (unsigned)(unsigned long long)p;
}

#if TDM_MODE
// Build a 2D tensor D# and issue TENSOR_LOAD_TO_LDS (ISA ch.8).
// dsz: data_size code (1 = 2 bytes). Dimensions in data_size units.
__device__ __forceinline__ void tdm_load_2d(unsigned lds_off, const void* gaddr,
                                            unsigned tile_x, unsigned tile_y,
                                            unsigned tensor_x, unsigned tensor_y,
                                            unsigned stride_x, unsigned dsz) {
  const unsigned long long ga = (unsigned long long)gaddr;
  u32x4 g0;
  g0[0] = 1u;                                              // count=1, user D#
  g0[1] = lds_off;                                         // lds_addr (bytes)
  g0[2] = (unsigned)(ga & 0xffffffffull);                  // global_addr[31:0]
  g0[3] = (unsigned)((ga >> 32) & 0x01ffffffull) | (2u << 30); // addr[56:32] | type=2
  i32x8 g1;
  g1[0] = (int)(dsz << 16);                                // data_size
  g1[1] = (int)((tensor_x & 0xffffu) << 16);               // tensor_dim0[15:0]
  g1[2] = (int)(((tensor_x >> 16) & 0xffffu) | ((tensor_y & 0xffffu) << 16));
  g1[3] = (int)(((tensor_y >> 16) & 0xffffu) | ((tile_x & 0xffffu) << 16));
  g1[4] = (int)(tile_y & 0xffffu);                         // tile_dim1 (tile_dim2=0)
  g1[5] = (int)stride_x;                                   // tensor_dim0_stride[31:0]
  g1[6] = 0;
  g1[7] = 0;
  i32x4 z4 = {0, 0, 0, 0};
#if TDM_MODE == 6
  i32x8 z8 = {0, 0, 0, 0, 0, 0, 0, 0};
  __builtin_amdgcn_tensor_load_to_lds(g0, g1, z4, z4, z8, 0);
#else
  __builtin_amdgcn_tensor_load_to_lds(g0, g1, z4, z4, 0);
#endif
}
#endif

// 16x64 @ 64x64 projection: Out = Ain @ W + bias, one 16-wide N tile per wave.
__device__ __forceinline__ void proj64(const _Float16* Ain, const _Float16* __restrict__ W,
                                       const float* __restrict__ Bv, _Float16* Out,
                                       int n0, int col, int rb, int lane) {
  v8f acc;
  const float bsc = Bv[n0 + col];
#pragma unroll
  for (int r = 0; r < 8; ++r) acc[r] = bsc;
  for (int k0 = 0; k0 < DD; k0 += 32) {
    v16h a  = frag_a_lds(Ain, DD, k0, lane);
    v16h bb = frag_b_h(W, DD, k0, n0, lane);
    acc = wmma_f16(a, bb, acc);
  }
#pragma unroll
  for (int r = 0; r < 8; ++r)
    Out[(r + rb) * DD + n0 + col] = (_Float16)acc[r];
}

// ---------------------------------------------------------------------------
// Kernel 0: f32 -> f16 weight conversion pre-pass
// ---------------------------------------------------------------------------
__global__ void cvt16_k(const float* __restrict__ s, _Float16* __restrict__ d, int n) {
  const int i = blockIdx.x * blockDim.x + threadIdx.x;
  if (i < n) d[i] = (_Float16)s[i];
}

// ---------------------------------------------------------------------------
// Kernel 1: 4-layer post-norm transformer over agents + edge projection.
// One block (4 waves, 128 threads) per batch element b (NB=128 blocks).
// Weights arrive pre-converted to f16; FFN2 weights stream through TDM->LDS.
// ---------------------------------------------------------------------------
__global__ __launch_bounds__(128) void tf_kernel(
    const float* __restrict__ indiv_us,
    const _Float16* __restrict__ wq, const float* __restrict__ bq,
    const _Float16* __restrict__ wk, const float* __restrict__ bk,
    const _Float16* __restrict__ wv, const float* __restrict__ bv,
    const _Float16* __restrict__ wo, const float* __restrict__ bo,
    const _Float16* __restrict__ w1, const float* __restrict__ b1,
    const _Float16* __restrict__ w2, const float* __restrict__ b2,
    const float* __restrict__ g1, const float* __restrict__ be1,
    const float* __restrict__ g2, const float* __restrict__ be2,
    const _Float16* __restrict__ edge_w, const float* __restrict__ edge_b,
    float* __restrict__ hg_out) {
  const int b    = blockIdx.x;
  const int tid  = threadIdx.x;
  const int wave = tid >> 5;
  const int lane = tid & 31;
  const int col  = lane & 15;
  const int rb   = (lane >> 4) << 3;   // C/D row base: +8 for hi half-wave
  const int n0   = wave * 16;          // each wave owns one 16-wide N tile

  __shared__ float    xs[NAg * DD];      // current x, f32 (LN precision)
  __shared__ _Float16 xh[NAg * DD];      // f16 mirror for WMMA A
  __shared__ _Float16 qsL[NAg * DD];
  __shared__ _Float16 ksL[NAg * DD];
  __shared__ _Float16 vsL[NAg * DD];
  __shared__ float    sc[NAg * NAg];     // attention logits
  __shared__ _Float16 ph[NAg * 32];      // softmax probs, K-padded to 32
  __shared__ _Float16 ao[NAg * DD];      // attn @ V
  __shared__ float    yb[NAg * DD];      // residual branch output
  __shared__ _Float16 h1[NAg * FFd];     // FFN hidden (64 KB)
  __shared__ _Float16 wstage[2][32 * DD]; // TDM double buffer for W2 chunks (8 KB)

  for (int i = tid; i < NAg * DD; i += 128) {
    float v = indiv_us[b * (NAg * DD) + i];
    xs[i] = v;
    xh[i] = (_Float16)v;
  }
  __syncthreads();

  for (int L = 0; L < NLr; ++L) {
    // ---- QKV projections
    proj64(xh, wq + L * DD * DD, bq + L * DD, qsL, n0, col, rb, lane);
    proj64(xh, wk + L * DD * DD, bk + L * DD, ksL, n0, col, rb, lane);
    proj64(xh, wv + L * DD * DD, bv + L * DD, vsL, n0, col, rb, lane);
    __syncthreads();

    // ---- scores = q @ k^T / sqrt(d)   (16x16, wave 0 only)
    if (wave == 0) {
      v8f acc = {0.f, 0.f, 0.f, 0.f, 0.f, 0.f, 0.f, 0.f};
      const int hb = (lane >> 4) << 4;
      for (int k0 = 0; k0 < DD; k0 += 32) {
        v16h a = frag_a_lds(qsL, DD, k0, lane);
        v16h bb;
#pragma unroll
        for (int e = 0; e < 16; ++e)            // B[kk][n] = K[n][kk]
          bb[e] = ksL[col * DD + k0 + e + hb];
        acc = wmma_f16(a, bb, acc);
      }
#pragma unroll
      for (int r = 0; r < 8; ++r)
        sc[(r + rb) * NAg + col] = acc[r] * 0.125f;   // 1/sqrt(64)
    }
    __syncthreads();

    // ---- softmax rows -> ph (zero-padded to K=32)
    if (tid < NAg) {
      float mx = -3.4e38f;
      for (int j = 0; j < NAg; ++j) mx = fmaxf(mx, sc[tid * NAg + j]);
      float s = 0.f;
      for (int j = 0; j < NAg; ++j) s += __expf(sc[tid * NAg + j] - mx);
      const float inv = 1.f / s;
      for (int j = 0; j < NAg; ++j) {
        ph[tid * 32 + j]      = (_Float16)(__expf(sc[tid * NAg + j] - mx) * inv);
        ph[tid * 32 + 16 + j] = (_Float16)0.f;
      }
    }
    __syncthreads();

    // ---- AO = P @ V  (K padded to 32, pad lanes masked to 0)
    {
      v8f acc = {0.f, 0.f, 0.f, 0.f, 0.f, 0.f, 0.f, 0.f};
      v16h a = frag_a_lds(ph, 32, 0, lane);
      const int hb = (lane >> 4) << 4;
      v16h bb;
#pragma unroll
      for (int e = 0; e < 16; ++e) {
        const int kk = e + hb;
        bb[e] = (kk < 16) ? vsL[kk * DD + n0 + col] : (_Float16)0.f;
      }
      acc = wmma_f16(a, bb, acc);
#pragma unroll
      for (int r = 0; r < 8; ++r)
        ao[(r + rb) * DD + n0 + col] = (_Float16)acc[r];
    }
    __syncthreads();

    // ---- Y = AO @ Wo + bo
    {
      const _Float16* Wo = wo + L * DD * DD;
      v8f acc;
      const float bsc = bo[L * DD + n0 + col];
#pragma unroll
      for (int r = 0; r < 8; ++r) acc[r] = bsc;
      for (int k0 = 0; k0 < DD; k0 += 32) {
        v16h a  = frag_a_lds(ao, DD, k0, lane);
        v16h bb = frag_b_h(Wo, DD, k0, n0, lane);
        acc = wmma_f16(a, bb, acc);
      }
#pragma unroll
      for (int r = 0; r < 8; ++r)
        yb[(r + rb) * DD + n0 + col] = acc[r];
    }
    __syncthreads();

    // ---- LN1: x = LN(x + Y) * g1 + be1
    if (tid < NAg) {
      const float* G  = g1  + L * DD;
      const float* BE = be1 + L * DD;
      float m = 0.f;
      for (int j = 0; j < DD; ++j) m += xs[tid * DD + j] + yb[tid * DD + j];
      m *= (1.f / DD);
      float vv = 0.f;
      for (int j = 0; j < DD; ++j) {
        float d = xs[tid * DD + j] + yb[tid * DD + j] - m;
        vv += d * d;
      }
      const float is = rsqrtf(vv * (1.f / DD) + 1e-5f);
      for (int j = 0; j < DD; ++j) {
        float val = (xs[tid * DD + j] + yb[tid * DD + j] - m) * is * G[j] + BE[j];
        xs[tid * DD + j] = val;
        xh[tid * DD + j] = (_Float16)val;
      }
    }
    __syncthreads();

    // ---- FFN1: h1 = relu(x @ W1 + b1), N = 2048 -> 32 tiles per wave
    {
      const _Float16* W1 = w1 + L * DD * FFd;
      const float* B1 = b1 + L * FFd;
      for (int nt = wave; nt < FFd / 16; nt += 4) {
        const int nn0 = nt * 16;
        __builtin_prefetch(&W1[nn0 + 1024], 0, 0);
        v8f acc;
        const float bsc = B1[nn0 + col];
#pragma unroll
        for (int r = 0; r < 8; ++r) acc[r] = bsc;
        for (int k0 = 0; k0 < DD; k0 += 32) {
          v16h a  = frag_a_lds(xh, DD, k0, lane);
          v16h bb = frag_b_h(W1, FFd, k0, nn0, lane);
          acc = wmma_f16(a, bb, acc);
        }
#pragma unroll
        for (int r = 0; r < 8; ++r)
          h1[(r + rb) * FFd + nn0 + col] = (_Float16)fmaxf(acc[r], 0.f);
      }
    }
    __syncthreads();

    // ---- FFN2: y = h1 @ W2 + b2, K = 2048, W2 chunks streamed TDM -> LDS
    {
      const _Float16* W2h = w2 + L * FFd * DD;
      v8f acc;
      const float bsc = b2[L * DD + n0 + col];
#pragma unroll
      for (int r = 0; r < 8; ++r) acc[r] = bsc;
      const int hb = (lane >> 4) << 4;
#if TDM_MODE
      if (wave == 0)   // prologue: stage chunk 0 (32 K-rows x 64 cols, f16)
        tdm_load_2d(lds_addr32(&wstage[0][0]), W2h, DD, 32, DD, FFd, DD, 1u);
#endif
      for (int c = 0; c < FFd / 32; ++c) {
        const int k0 = c * 32;
#if TDM_MODE
        if (wave == 0) {
          if (c + 1 < FFd / 32) {
            tdm_load_2d(lds_addr32(&wstage[(c + 1) & 1][0]),
                        W2h + (k0 + 32) * DD, DD, 32, DD, FFd, DD, 1u);
            __builtin_amdgcn_s_wait_tensorcnt(1);   // chunk c is resident
          } else {
            __builtin_amdgcn_s_wait_tensorcnt(0);
          }
        }
        __syncthreads();
#else
        for (int i = tid; i < 32 * DD; i += 128)
          wstage[c & 1][i] = W2h[k0 * DD + i];
        __syncthreads();
#endif
        const _Float16* wb = wstage[c & 1];
        v16h a = frag_a_lds(h1, FFd, k0, lane);
        v16h bb;
#pragma unroll
        for (int e = 0; e < 16; ++e)
          bb[e] = wb[(e + hb) * DD + n0 + col];
        acc = wmma_f16(a, bb, acc);
        __syncthreads();   // buffer free before it is re-filled
      }
#pragma unroll
      for (int r = 0; r < 8; ++r)
        yb[(r + rb) * DD + n0 + col] = acc[r];
    }
    __syncthreads();

    // ---- LN2
    if (tid < NAg) {
      const float* G  = g2  + L * DD;
      const float* BE = be2 + L * DD;
      float m = 0.f;
      for (int j = 0; j < DD; ++j) m += xs[tid * DD + j] + yb[tid * DD + j];
      m *= (1.f / DD);
      float vv = 0.f;
      for (int j = 0; j < DD; ++j) {
        float d = xs[tid * DD + j] + yb[tid * DD + j] - m;
        vv += d * d;
      }
      const float is = rsqrtf(vv * (1.f / DD) + 1e-5f);
      for (int j = 0; j < DD; ++j) {
        float val = (xs[tid * DD + j] + yb[tid * DD + j] - m) * is * G[j] + BE[j];
        xs[tid * DD + j] = val;
        xh[tid * DD + j] = (_Float16)val;
      }
    }
    __syncthreads();
  }

  // ---- hyper_graph = relu(x @ edge_w + edge_b), 16x16 (wave 0)
  if (wave == 0) {
    v8f acc;
    const float bsc = edge_b[col];
#pragma unroll
    for (int r = 0; r < 8; ++r) acc[r] = bsc;
    for (int k0 = 0; k0 < DD; k0 += 32) {
      v16h a  = frag_a_lds(xh, DD, k0, lane);
      v16h bb = frag_b_h(edge_w, NEd, k0, 0, lane);
      acc = wmma_f16(a, bb, acc);
    }
#pragma unroll
    for (int r = 0; r < 8; ++r)
      hg_out[b * (NAg * NEd) + (r + rb) * NEd + col] = fmaxf(acc[r], 0.f);
  }
}

// ---------------------------------------------------------------------------
// Kernel 2a: WMMA GEMM, C = act(A[M,K](f32) @ B[K,N](f32) + bias)
// ---------------------------------------------------------------------------
__global__ __launch_bounds__(32) void gemm_k(
    const float* __restrict__ A, int lda,
    const float* __restrict__ Bm, int ldb,
    const float* __restrict__ bias,
    float* __restrict__ C, int ldc, int K, int act) {
  const int lane = threadIdx.x & 31;
  const int n0   = blockIdx.x * 16;
  const int m0   = blockIdx.y * 16;
  const int col  = lane & 15;
  const int row  = lane & 15;
  const int hbA  = (lane >> 4) << 4;
  const int hbB  = (lane >> 4) << 4;
  const int rb   = (lane >> 4) << 3;

  v8f acc;
  const float bsc = bias[n0 + col];
#pragma unroll
  for (int r = 0; r < 8; ++r) acc[r] = bsc;

  for (int k0 = 0; k0 < K; k0 += 32) {
    v16h a;
#pragma unroll
    for (int e = 0; e < 16; ++e) {
      int kk = e + hbA;
      a[e] = (_Float16)A[(m0 + row) * lda + k0 + kk];
    }
    v16h b;
#pragma unroll
    for (int e = 0; e < 16; ++e)
      b[e] = (_Float16)Bm[(k0 + e + hbB) * ldb + n0 + col];
    acc = wmma_f16(a, b, acc);
  }
#pragma unroll
  for (int r = 0; r < 8; ++r) {
    float v = acc[r];
    if (act == 1)      v = fmaxf(v, 0.f);
    else if (act == 2) v = fabsf(v);
    C[(m0 + r + rb) * ldc + n0 + col] = v;
  }
}

// ---------------------------------------------------------------------------
// Kernel 2b: WMMA GEMM with pre-converted f16 B (used for the big hc1_w2)
// ---------------------------------------------------------------------------
__global__ __launch_bounds__(32) void gemm_kh(
    const float* __restrict__ A, int lda,
    const _Float16* __restrict__ Bm, int ldb,
    const float* __restrict__ bias,
    float* __restrict__ C, int ldc, int K, int act) {
  const int lane = threadIdx.x & 31;
  const int n0   = blockIdx.x * 16;
  const int m0   = blockIdx.y * 16;
  const int col  = lane & 15;
  const int row  = lane & 15;
  const int hbA  = (lane >> 4) << 4;
  const int hbB  = (lane >> 4) << 4;
  const int rb   = (lane >> 4) << 3;

  v8f acc;
  const float bsc = bias[n0 + col];
#pragma unroll
  for (int r = 0; r < 8; ++r) acc[r] = bsc;

  for (int k0 = 0; k0 < K; k0 += 32) {
    v16h a;
#pragma unroll
    for (int e = 0; e < 16; ++e) {
      int kk = e + hbA;
      a[e] = (_Float16)A[(m0 + row) * lda + k0 + kk];
    }
    v16h b;
#pragma unroll
    for (int e = 0; e < 16; ++e)
      b[e] = Bm[(k0 + e + hbB) * ldb + n0 + col];
    acc = wmma_f16(a, b, acc);
  }
#pragma unroll
  for (int r = 0; r < 8; ++r) {
    float v = acc[r];
    if (act == 1)      v = fmaxf(v, 0.f);
    else if (act == 2) v = fabsf(v);
    C[(m0 + r + rb) * ldc + n0 + col] = v;
  }
}

// ---------------------------------------------------------------------------
// Kernel 3: new_hg[bs,a,e] = max_j( hg[b,a,gc[bs,e,j]] * map[bs,e,j] )
// ---------------------------------------------------------------------------
__global__ void newhg_k(const float* __restrict__ hg,
                        const int* __restrict__ gc,
                        const int* __restrict__ smap,
                        float* __restrict__ nh) {
  const int idx = blockIdx.x * blockDim.x + threadIdx.x;
  if (idx >= NB * NS * NAg * NEd) return;
  const int bs  = idx >> 8;
  const int rem = idx & 255;
  const int a   = rem >> 4;
  const int e   = rem & 15;
  const int b   = bs >> 5;
  const int base = bs * (NEd * NEd) + e * NEd;
  const float* hrow = hg + b * (NAg * NEd) + a * NEd;
  float mx = -3.4e38f;
#pragma unroll
  for (int j = 0; j < NEd; ++j) {
    const int ci = gc[base + j];
    const float v = hrow[ci] * (float)smap[base + j];
    mx = fmaxf(mx, v);
  }
  nh[bs * (NAg * NEd) + a * NEd + e] = mx;
}

// ---------------------------------------------------------------------------
// Kernel 4: marginal utility -> qs_tot X[bs,e]. One wave per coalition sample.
// ---------------------------------------------------------------------------
__global__ __launch_bounds__(32) void margin_k(const float* __restrict__ nh,
                                               const float* __restrict__ wabs,
                                               const float* __restrict__ rest,
                                               float* __restrict__ X) {
  const int bs = blockIdx.x;
  const int b  = bs >> 5;
  const int t  = threadIdx.x;
  __shared__ float tile[NAg * NEd];
  __shared__ float binv[NEd];
  __shared__ float dinv[NAg];
  for (int i = t; i < NAg * NEd; i += 32) tile[i] = nh[bs * (NAg * NEd) + i];
  __syncthreads();
  if (t < 16) {
    float cs = 0.f;
    for (int a = 0; a < NAg; ++a) cs += tile[a * NEd + t];
    binv[t] = (cs != 0.f) ? (1.f / cs) : 0.f;
  } else {
    const int a = t - 16;
    float dg = 0.f;
    for (int e = 0; e < NEd; ++e) dg += tile[a * NEd + e] * wabs[b * NEd + e];
    dinv[a] = (dg > 0.f) ? rsqrtf(dg) : 0.f;
  }
  __syncthreads();
  if (t < 16) {
    float x = 0.f;
    for (int a = 0; a < NAg; ++a)
      x += dinv[a] * tile[a * NEd + t] * rest[b * NAg + a];
    X[bs * NEd + t] = x * wabs[b * NEd + t] * binv[t];
  }
}

// ---------------------------------------------------------------------------
// Kernel 5: mixing head. One block per coalition sample, one thread per k.
// ---------------------------------------------------------------------------
__global__ __launch_bounds__(256) void mix_k(const float* __restrict__ Xb,
                                             const float* __restrict__ agent_qs,
                                             const float* __restrict__ hw1b,
                                             const float* __restrict__ hc1b,
                                             const float* __restrict__ hwwb,
                                             const float* __restrict__ hccb,
                                             float* __restrict__ out) {
  const int bs = blockIdx.x;
  const int b  = bs >> 5;
  const int k  = threadIdx.x;           // k = a*16 + e
  const int a  = k >> 4;
  const int e  = k & 15;
  __shared__ float w0s[HH], w1s[HH], wws[HH];
  if (k < HH) {
    w0s[k] = hw1b[b * (2 * HH) + k];
    w1s[k] = hw1b[b * (2 * HH) + HH + k];
    wws[k] = hwwb[b * HH + k];
  }
  __syncthreads();
  const float x0 = Xb[bs * NEd + e];
  const float x1 = agent_qs[b * NAg + a];
  const float* hc = hc1b + (b * (NAg * NEd) + k) * HH;
  float acc = 0.f;
  for (int h = 0; h < HH; ++h) {
    float v  = (x0 * w0s[h] + x1 * w1s[h]) * 0.70710678118654752f + hc[h];
    float el = (v > 0.f) ? v : (__expf(v) - 1.f);
    acc += el * wws[h];
  }
  const float qt = acc * 0.125f + hccb[b * (NAg * NEd) + k];  // 1/sqrt(64)
  out[bs * (NAg * NEd) + k] = fmaxf(-qt, 0.f);
}

// ---------------------------------------------------------------------------
extern "C" void kernel_launch(void* const* d_in, const int* in_sizes, int n_in,
                              void* d_out, int out_size, void* d_ws, size_t ws_size,
                              hipStream_t stream) {
  const float* agent_qs = (const float*)d_in[0];
  const float* states   = (const float*)d_in[1];
  const float* indiv_us = (const float*)d_in[2];
  const float* rest     = (const float*)d_in[3];
  const int*   smap     = (const int*)d_in[4];
  const int*   gc       = (const int*)d_in[5];
  const float* tf_wq = (const float*)d_in[6];
  const float* tf_bq = (const float*)d_in[7];
  const float* tf_wk = (const float*)d_in[8];
  const float* tf_bk = (const float*)d_in[9];
  const float* tf_wv = (const float*)d_in[10];
  const float* tf_bv = (const float*)d_in[11];
  const float* tf_wo = (const float*)d_in[12];
  const float* tf_bo = (const float*)d_in[13];
  const float* tf_w1 = (const float*)d_in[14];
  const float* tf_b1 = (const float*)d_in[15];
  const float* tf_w2 = (const float*)d_in[16];
  const float* tf_b2 = (const float*)d_in[17];
  const float* tf_g1 = (const float*)d_in[18];
  const float* tf_be1 = (const float*)d_in[19];
  const float* tf_g2 = (const float*)d_in[20];
  const float* tf_be2 = (const float*)d_in[21];
  const float* edge_w = (const float*)d_in[22];
  const float* edge_b = (const float*)d_in[23];
  const float* wl_w1 = (const float*)d_in[24];
  const float* wl_b1 = (const float*)d_in[25];
  const float* wl_w2 = (const float*)d_in[26];
  const float* wl_b2 = (const float*)d_in[27];
  const float* hw1_w1 = (const float*)d_in[28];
  const float* hw1_b1 = (const float*)d_in[29];
  const float* hw1_w2 = (const float*)d_in[30];
  const float* hw1_b2 = (const float*)d_in[31];
  const float* hc1_w1 = (const float*)d_in[32];
  const float* hc1_b1 = (const float*)d_in[33];
  const float* hc1_w2 = (const float*)d_in[34];
  const float* hc1_b2 = (const float*)d_in[35];
  const float* hw_w1 = (const float*)d_in[36];
  const float* hw_b1 = (const float*)d_in[37];
  const float* hw_w2 = (const float*)d_in[38];
  const float* hw_b2 = (const float*)d_in[39];
  const float* hc_w1 = (const float*)d_in[40];
  const float* hc_b1 = (const float*)d_in[41];
  const float* hc_w2 = (const float*)d_in[42];
  const float* hc_b2 = (const float*)d_in[43];
  (void)in_sizes; (void)n_in; (void)out_size; (void)ws_size;

  // Workspace carve-up: f32 region, then f16 region
  float* W     = (float*)d_ws;
  float* hgb   = W;                    // 128*16*16      = 32768
  float* nhb   = hgb  + 32768;         // 4096*16*16     = 1048576
  float* wabsb = nhb  + 1048576;       // 128*16         = 2048
  float* Xb    = wabsb + 2048;         // 4096*16        = 65536
  float* Tb    = Xb   + 65536;         // up to 128*256  = 32768
  float* hw1b  = Tb   + 32768;         // 128*128        = 16384
  float* hc1b  = hw1b + 16384;         // 128*16384      = 2097152
  float* hwwb  = hc1b + 2097152;       // 128*64         = 8192
  float* hccb  = hwwb + 8192;          // 128*256        = 32768

  _Float16* Hbase = (_Float16*)(hccb + 32768);
  _Float16* wqh   = Hbase;             // 4*64*64  = 16384
  _Float16* wkh   = wqh + 16384;
  _Float16* wvh   = wkh + 16384;
  _Float16* woh   = wvh + 16384;
  _Float16* w1h   = woh + 16384;       // 4*64*2048 = 524288
  _Float16* w2h   = w1h + 524288;      // 4*2048*64 = 524288
  _Float16* ewh   = w2h + 524288;      // 64*16     = 1024
  _Float16* hc1w2h = ewh + 1024;       // 128*16384 = 2097152

  const float* st = states;            // [128, 128]

  // 0) f32 -> f16 weight pre-conversion
  cvt16_k<<<(16384 + 255) / 256, 256, 0, stream>>>(tf_wq, wqh, 16384);
  cvt16_k<<<(16384 + 255) / 256, 256, 0, stream>>>(tf_wk, wkh, 16384);
  cvt16_k<<<(16384 + 255) / 256, 256, 0, stream>>>(tf_wv, wvh, 16384);
  cvt16_k<<<(16384 + 255) / 256, 256, 0, stream>>>(tf_wo, woh, 16384);
  cvt16_k<<<(524288 + 255) / 256, 256, 0, stream>>>(tf_w1, w1h, 524288);
  cvt16_k<<<(524288 + 255) / 256, 256, 0, stream>>>(tf_w2, w2h, 524288);
  cvt16_k<<<(1024 + 255) / 256, 256, 0, stream>>>(edge_w, ewh, 1024);
  cvt16_k<<<(2097152 + 255) / 256, 256, 0, stream>>>(hc1_w2, hc1w2h, 2097152);

  // 1) transformer + edge -> hyper_graph
  tf_kernel<<<NB, 128, 0, stream>>>(indiv_us, wqh, tf_bq, wkh, tf_bk,
                                    wvh, tf_bv, woh, tf_bo,
                                    w1h, tf_b1, w2h, tf_b2,
                                    tf_g1, tf_be1, tf_g2, tf_be2,
                                    ewh, edge_b, hgb);

  // 2) state hypernetworks (all WMMA GEMMs, M=128)
  gemm_k<<<dim3(8, 8),    32, 0, stream>>>(st, 128, wl_w1, 128, wl_b1, Tb, 128, 128, 1);
  gemm_k<<<dim3(1, 8),    32, 0, stream>>>(Tb, 128, wl_w2, 16, wl_b2, wabsb, 16, 128, 2);
  gemm_k<<<dim3(12, 8),   32, 0, stream>>>(st, 128, hw1_w1, 192, hw1_b1, Tb, 192, 128, 1);
  gemm_k<<<dim3(8, 8),    32, 0, stream>>>(Tb, 192, hw1_w2, 128, hw1_b2, hw1b, 128, 192, 0);
  gemm_k<<<dim3(8, 8),    32, 0, stream>>>(st, 128, hc1_w1, 128, hc1_b1, Tb, 128, 128, 1);
  gemm_kh<<<dim3(1024, 8), 32, 0, stream>>>(Tb, 128, hc1w2h, 16384, hc1_b2, hc1b, 16384, 128, 0);
  gemm_k<<<dim3(4, 8),    32, 0, stream>>>(st, 128, hw_w1, 64, hw_b1, Tb, 64, 128, 1);
  gemm_k<<<dim3(4, 8),    32, 0, stream>>>(Tb, 64, hw_w2, 64, hw_b2, hwwb, 64, 64, 2);
  gemm_k<<<dim3(4, 8),    32, 0, stream>>>(st, 128, hc_w1, 64, hc_b1, Tb, 64, 128, 1);
  gemm_k<<<dim3(16, 8),   32, 0, stream>>>(Tb, 64, hc_w2, 256, hc_b2, hccb, 256, 64, 0);

  // 3) coalition gather/mask/max
  newhg_k<<<(NB * NS * NAg * NEd + 255) / 256, 256, 0, stream>>>(hgb, gc, smap, nhb);

  // 4) marginal utility -> qs_tot
  margin_k<<<NB * NS, 32, 0, stream>>>(nhb, wabsb, rest, Xb);

  // 5) mixing head -> relu(-qt)
  mix_k<<<NB * NS, 256, 0, stream>>>(Xb, agent_qs, hw1b, hc1b, hwwb, hccb,
                                     (float*)d_out);
}
